// SpatialTransformer3D_44435731644573
// MI455X (gfx1250) — matrix operand order repaired and checked by reading, not compiled, not run on hardware
//
#include <hip/hip_runtime.h>

// ---------------------------------------------------------------------------
// Windowed causal attention, all matmuls on v_wmma_f32_16x16x32_f16 (gfx1250)
// B=2 F=8 H=W=32 C=512 ws=8  ->  N=32 windows, S=512, heads=8, dh=64
// ---------------------------------------------------------------------------

#define WIN_N 32
#define SEQ   512
#define CD    512
#define MTOT  (WIN_N * SEQ)   // 16384 rows

typedef __attribute__((ext_vector_type(16))) _Float16 v16h;
typedef __attribute__((ext_vector_type(8)))  float    v8f;
typedef int v4i __attribute__((vector_size(16)));      // matches builtin param

union AF { v16h v; unsigned int u[8]; unsigned short s[16]; _Float16 h[16]; };
union F8 { v8f v; float f[8]; };

// ---- optional CDNA5 async global->LDS path (guarded; fallback = reg-staged)
#if defined(__has_builtin)
#  if __has_builtin(__builtin_amdgcn_global_load_async_to_lds_b128)
#    define ASYNC_LDS 1
#  endif
#endif
#ifndef ASYNC_LDS
#  define ASYNC_LDS 0
#endif

#if ASYNC_LDS
static __device__ __forceinline__ void async_cp16(void* lds, const void* g) {
  __builtin_amdgcn_global_load_async_to_lds_b128(
      (__attribute__((address_space(1))) v4i*)(uintptr_t)g,
      (__attribute__((address_space(3))) v4i*)(unsigned int)(uintptr_t)lds,
      0, 0);
}
static __device__ __forceinline__ void async_wait0() {
#  if __has_builtin(__builtin_amdgcn_s_wait_asynccnt)
  __builtin_amdgcn_s_wait_asynccnt(0);
#  else
  asm volatile("s_wait_asynccnt 0x0" ::: "memory");
#  endif
}
#endif

static __device__ __forceinline__ v8f wmma32f16(v16h a, v16h b, v8f c) {
  // (neg_a, A, neg_b, B, c_mod, C, reuse_a, reuse_b)
  return __builtin_amdgcn_wmma_f32_16x16x32_f16(false, a, false, b, (short)0, c,
                                                false, false);
}

static __device__ __forceinline__ unsigned short f2h(float x) {
  union { _Float16 h; unsigned short u; } cv; cv.h = (_Float16)x; return cv.u;
}

static __device__ __forceinline__ void put8(AF& a, uint4 x, uint4 y) {
  a.u[0] = x.x; a.u[1] = x.y; a.u[2] = x.z; a.u[3] = x.w;
  a.u[4] = y.x; a.u[5] = y.y; a.u[6] = y.z; a.u[7] = y.w;
}

// A-fragment (16x32 f16, M x K): lane lr holds row lr; K slot for vgpr v,
// half = lane>>4:  k = half*8 + (v&3)*2 + (v>=4 ? 16 : 0)   [ISA 7.12.2]
static __device__ __forceinline__ int a_k(int v, int half) {
  return half * 8 + (v & 3) * 2 + ((v >= 4) ? 16 : 0);
}

// ---------------------------------------------------------------------------
// 1) fp32 x -> f16 windowed (N, S, C)
// ---------------------------------------------------------------------------
__global__ __launch_bounds__(256)
void pack_win(const float* __restrict__ x, _Float16* __restrict__ winh) {
  int t   = blockIdx.x * 256 + threadIdx.x;      // 2,097,152 threads
  int row = t >> 7;                              // 0..16383
  int c   = (t & 127) << 2;                      // 0..508
  int n = row >> 9, s = row & 511;
  int b = n & 1, wb = (n >> 1) & 3, hb = n >> 3;
  int f = s >> 6, r8 = (s >> 3) & 7, c8 = s & 7;
  long xi = (((long)(b * 8 + f) * 32 + (hb * 8 + r8)) * 32 + (wb * 8 + c8)) * 512 + c;
  float4 vv = *(const float4*)(x + xi);
  union { _Float16 h[4]; uint2 u; } o;
  o.h[0] = (_Float16)vv.x; o.h[1] = (_Float16)vv.y;
  o.h[2] = (_Float16)vv.z; o.h[3] = (_Float16)vv.w;
  *(uint2*)(winh + (long)row * 512 + c) = o.u;
}

// ---------------------------------------------------------------------------
// 2) weights fp32 -> f16 (blockIdx.y selects matrix)
// ---------------------------------------------------------------------------
__global__ __launch_bounds__(256)
void conv_w(const float* __restrict__ w0, const float* __restrict__ w1,
            const float* __restrict__ w2, const float* __restrict__ w3,
            _Float16* __restrict__ o0, _Float16* __restrict__ o1,
            _Float16* __restrict__ o2, _Float16* __restrict__ o3) {
  const float* src[4] = {w0, w1, w2, w3};
  _Float16*    dst[4] = {o0, o1, o2, o3};
  int m = blockIdx.y;
  int i = (blockIdx.x * 256 + threadIdx.x) * 4;  // 262144 elems per matrix
  float4 vv = *(const float4*)(src[m] + i);
  union { _Float16 h[4]; uint2 u; } o;
  o.h[0] = (_Float16)vv.x; o.h[1] = (_Float16)vv.y;
  o.h[2] = (_Float16)vv.z; o.h[3] = (_Float16)vv.w;
  *(uint2*)(dst[m] + i) = o.u;
}

// ---------------------------------------------------------------------------
// Shared GEMM tile-staging step: stage A(128x32) and B(32x64) into LDS
// ---------------------------------------------------------------------------
static __device__ __forceinline__ void stage_tiles(
    const _Float16* __restrict__ A, const _Float16* __restrict__ W,
    unsigned short (*ldsA)[40], unsigned short (*ldsB)[72],
    int mBase, int cb, int kk, int ar, int aseg, int br, int bseg) {
#if ASYNC_LDS
  __syncthreads();                               // prior reads complete
  async_cp16(&ldsA[ar][aseg],      A + (long)(mBase + ar) * 512 + kk + aseg);
  async_cp16(&ldsA[ar + 64][aseg], A + (long)(mBase + ar + 64) * 512 + kk + aseg);
  async_cp16(&ldsB[br][bseg],      W + (long)(kk + br) * 512 + cb + bseg);
  async_wait0();
  __syncthreads();
#else
  uint4 av0 = *(const uint4*)(A + (long)(mBase + ar) * 512 + kk + aseg);
  uint4 av1 = *(const uint4*)(A + (long)(mBase + ar + 64) * 512 + kk + aseg);
  uint4 bv  = *(const uint4*)(W + (long)(kk + br) * 512 + cb + bseg);
  __syncthreads();
  *(uint4*)&ldsA[ar][aseg]      = av0;
  *(uint4*)&ldsA[ar + 64][aseg] = av1;
  *(uint4*)&ldsB[br][bseg]      = bv;
  __syncthreads();
#endif
  if (kk + 32 < 512) {                           // prefetch next K tiles to L2
    __builtin_prefetch(A + (long)(mBase + ar) * 512 + kk + 32 + aseg, 0, 3);
    __builtin_prefetch(A + (long)(mBase + ar + 64) * 512 + kk + 32 + aseg, 0, 3);
    __builtin_prefetch(W + (long)(kk + 32 + br) * 512 + cb + bseg, 0, 3);
  }
}

// ---------------------------------------------------------------------------
// 3) QKV GEMM: (16384 x 512) @ (512 x 512) f16 -> (N,h,S,dh) f16, opt. rotary
//    128x64 tile per WG; 8 waves, each 32x32 (4 accumulators, 4 WMMA/K-step)
// ---------------------------------------------------------------------------
__global__ __launch_bounds__(256)
void gemm_qkv(const _Float16* __restrict__ A, const _Float16* __restrict__ W,
              _Float16* __restrict__ out, int rotary) {
  __shared__ unsigned short ldsA[128][40];  // 128 rows x 32 K (+pad)
  __shared__ unsigned short ldsB[32][72];   // 32 K x 64 N (+pad)
  const int tid  = threadIdx.x;
  const int w    = tid >> 5, l = tid & 31;
  const int wm   = w >> 1, wn = w & 1;      // wave tile: rows wm*32, cols wn*32
  const int half = l >> 4, lr = l & 15;
  const int mBase = blockIdx.x * 128;
  const int cb    = blockIdx.y * 64;

  v8f acc00 = {}, acc01 = {}, acc10 = {}, acc11 = {};
  const int ar = tid >> 2, aseg = (tid & 3) * 8;   // A: 2 rows per thread
  const int br = tid >> 3, bseg = (tid & 7) * 8;

  for (int kk = 0; kk < 512; kk += 32) {
    stage_tiles(A, W, ldsA, ldsB, mBase, cb, kk, ar, aseg, br, bseg);

    AF a0, a1;
#pragma unroll
    for (int v = 0; v < 8; ++v) {
      a0.u[v] = *(const unsigned int*)&ldsA[wm * 32 + lr][a_k(v, half)];
      a1.u[v] = *(const unsigned int*)&ldsA[wm * 32 + 16 + lr][a_k(v, half)];
    }
    AF b0, b1;
#pragma unroll
    for (int s = 0; s < 16; ++s) {
      const int kr = a_k(s >> 1, half) + (s & 1);
      b0.s[s] = ldsB[kr][wn * 32 + lr];
      b1.s[s] = ldsB[kr][wn * 32 + 16 + lr];
    }

    acc00 = wmma32f16(a0.v, b0.v, acc00);
    acc01 = wmma32f16(a0.v, b1.v, acc01);
    acc10 = wmma32f16(a1.v, b0.v, acc10);
    acc11 = wmma32f16(a1.v, b1.v, acc11);
  }

  F8 res[2][2];
  res[0][0].v = acc00; res[0][1].v = acc01;
  res[1][0].v = acc10; res[1][1].v = acc11;
#pragma unroll
  for (int mi = 0; mi < 2; ++mi)
#pragma unroll
    for (int i = 0; i < 8; ++i) {
      int M = i + half * 8;
      int rGlob = mBase + wm * 32 + mi * 16 + M;
      int n = rGlob >> 9, s = rGlob & 511;
#pragma unroll
      for (int nt = 0; nt < 2; ++nt) {
        float val = res[mi][nt].f[i];
        int c = cb + wn * 32 + nt * 16 + lr;
        float partner = __shfl_xor(val, 1);    // neighbor channel (same row)
        int dhi = c & 63;
        if (rotary && dhi < 32) {
          // inv_freq = 10000^(-(dhi&~1)/32); ln(10000)=9.21034037
          float fr = __expf(-(float)(dhi & ~1) * (9.210340371976184f / 32.0f));
          float sn, cs; __sincosf((float)s * fr, &sn, &cs);
          val = (dhi & 1) ? (val * cs + partner * sn) : (val * cs - partner * sn);
        }
        int hd = c >> 6;
        out[((long)(n * 8 + hd) * 512 + s) * 64 + dhi] = (_Float16)val;
      }
    }
}

// ---------------------------------------------------------------------------
// 4) Flash-style causal attention per (window*head, 128-row block)
//    8 waves x 16 rows; KV chunks of 32; 8 WMMAs per chunk per wave
// ---------------------------------------------------------------------------
__global__ __launch_bounds__(256)
void attn(const _Float16* __restrict__ Q, const _Float16* __restrict__ Kb,
          const _Float16* __restrict__ Vb, _Float16* __restrict__ O) {
  __shared__ unsigned short ldsKT[64][40];     // K chunk transposed: [dh][kv]
  __shared__ unsigned short ldsV[32][72];      // V chunk: [kv][dh]
  __shared__ unsigned short ldsP[8][16][40];   // per-wave P tile 16x32
  const int tid  = threadIdx.x;
  const int w    = tid >> 5, l = tid & 31;
  const int half = l >> 4, lr = l & 15;
  const int rb   = blockIdx.x;
  const int nh   = blockIdx.y;                 // n*8 + head
  const long base = (long)nh * SEQ * 64;
  const int q0   = rb * 128 + w * 16;

  AF qf[2];                                    // Q 16x64 as two A-frags
#pragma unroll
  for (int f = 0; f < 2; ++f)
#pragma unroll
    for (int v = 0; v < 8; ++v)
      qf[f].u[v] = *(const unsigned int*)(Q + base + (long)(q0 + lr) * 64 +
                                          f * 32 + a_k(v, half));

  F8 oacc[4];
  float mst[8], lst[8];
  v8f zero = {};
#pragma unroll
  for (int dt = 0; dt < 4; ++dt) oacc[dt].v = zero;
#pragma unroll
  for (int i = 0; i < 8; ++i) { mst[i] = -3.0e38f; lst[i] = 0.0f; }

  const int krow = tid >> 3, seg = (tid & 7) * 8;
  const int kr = lr + half * 16;
  const int nchunks = rb * 4 + 4;              // causal: kv <= rb*128+127

  for (int ch = 0; ch < nchunks; ++ch) {
    const int t0 = ch * 32;
    uint4 kv4 = *(const uint4*)(Kb + base + (long)(t0 + krow) * 64 + seg);
    __syncthreads();
    union { uint4 q; unsigned short s[8]; } tk; tk.q = kv4;
#pragma unroll
    for (int j = 0; j < 8; ++j) ldsKT[seg + j][krow] = tk.s[j];   // transpose
#if ASYNC_LDS
    async_cp16(&ldsV[krow][seg], Vb + base + (long)(t0 + krow) * 64 + seg);
    async_wait0();
#else
    uint4 vv4 = *(const uint4*)(Vb + base + (long)(t0 + krow) * 64 + seg);
    *(uint4*)&ldsV[krow][seg] = vv4;
#endif
    __syncthreads();

    if (t0 <= q0 + 15) {                       // wave-uniform causal skip
      AF bk[2][2];                             // [dh-chunk f][score col tile]
#pragma unroll
      for (int f = 0; f < 2; ++f)
#pragma unroll
        for (int ct = 0; ct < 2; ++ct)
#pragma unroll
          for (int s = 0; s < 16; ++s)
            bk[f][ct].s[s] = ldsKT[f * 32 + a_k(s >> 1, half) + (s & 1)][ct * 16 + lr];

      F8 sc[2];
#pragma unroll
      for (int ct = 0; ct < 2; ++ct) {
        v8f a = zero;
        a = wmma32f16(qf[0].v, bk[0][ct].v, a);
        a = wmma32f16(qf[1].v, bk[1][ct].v, a);
        sc[ct].v = a;
      }
      // scale + causal mask
#pragma unroll
      for (int i = 0; i < 8; ++i) {
        int qrow = q0 + i + half * 8;
#pragma unroll
        for (int ct = 0; ct < 2; ++ct) {
          int kvc = t0 + ct * 16 + lr;
          float xv = sc[ct].f[i] * 0.125f;     // dh^-0.5
          sc[ct].f[i] = (kvc <= qrow) ? xv : -3.0e38f;
        }
      }
      // online softmax
#pragma unroll
      for (int i = 0; i < 8; ++i) {
        float mx = fmaxf(sc[0].f[i], sc[1].f[i]);
#pragma unroll
        for (int d = 1; d < 16; d <<= 1) mx = fmaxf(mx, __shfl_xor(mx, d));
        float mnew  = fmaxf(mst[i], mx);
        float scale = __expf(mst[i] - mnew);
        float p0 = __expf(sc[0].f[i] - mnew);
        float p1 = __expf(sc[1].f[i] - mnew);
        sc[0].f[i] = p0; sc[1].f[i] = p1;
        float rs = p0 + p1;
#pragma unroll
        for (int d = 1; d < 16; d <<= 1) rs += __shfl_xor(rs, d);
        lst[i] = lst[i] * scale + rs;
        mst[i] = mnew;
#pragma unroll
        for (int dt = 0; dt < 4; ++dt) oacc[dt].f[i] *= scale;
      }
      // P (C layout) -> LDS -> A-fragment
#pragma unroll
      for (int i = 0; i < 8; ++i) {
        int M = i + half * 8;
        ldsP[w][M][lr]      = f2h(sc[0].f[i]);
        ldsP[w][M][16 + lr] = f2h(sc[1].f[i]);
      }
      AF pf;
#pragma unroll
      for (int v = 0; v < 8; ++v)
        pf.u[v] = *(const unsigned int*)&ldsP[w][lr][a_k(v, half)];
      // P @ V
#pragma unroll
      for (int dt = 0; dt < 4; ++dt) {
        AF bv;
#pragma unroll
        for (int s = 0; s < 16; ++s)
          bv.s[s] = ldsV[a_k(s >> 1, half) + (s & 1)][dt * 16 + lr];
        oacc[dt].v = wmma32f16(pf.v, bv.v, oacc[dt].v);
      }
    }
    __syncthreads();
  }

  const int n = nh >> 3, hd = nh & 7;
#pragma unroll
  for (int i = 0; i < 8; ++i) {
    float inv = 1.0f / lst[i];
    int s = q0 + i + half * 8;
#pragma unroll
    for (int dt = 0; dt < 4; ++dt) {
      long oi = ((long)(n * 512 + s) * 512) + hd * 64 + dt * 16 + lr;
      O[oi] = (_Float16)(oacc[dt].f[i] * inv);
    }
  }
}

// ---------------------------------------------------------------------------
// 5) Output GEMM + bias, fp32 store in window-reversed (= x) layout
//    128x64 tile per WG; 32x32 per wave
// ---------------------------------------------------------------------------
__global__ __launch_bounds__(256)
void gemm_out(const _Float16* __restrict__ A, const _Float16* __restrict__ W,
              const float* __restrict__ bo, float* __restrict__ out) {
  __shared__ unsigned short ldsA[128][40];
  __shared__ unsigned short ldsB[32][72];
  const int tid  = threadIdx.x;
  const int w    = tid >> 5, l = tid & 31;
  const int wm   = w >> 1, wn = w & 1;
  const int half = l >> 4, lr = l & 15;
  const int mBase = blockIdx.x * 128;
  const int cb    = blockIdx.y * 64;

  v8f acc00 = {}, acc01 = {}, acc10 = {}, acc11 = {};
  const int ar = tid >> 2, aseg = (tid & 3) * 8;
  const int br = tid >> 3, bseg = (tid & 7) * 8;

  for (int kk = 0; kk < 512; kk += 32) {
    stage_tiles(A, W, ldsA, ldsB, mBase, cb, kk, ar, aseg, br, bseg);

    AF a0, a1;
#pragma unroll
    for (int v = 0; v < 8; ++v) {
      a0.u[v] = *(const unsigned int*)&ldsA[wm * 32 + lr][a_k(v, half)];
      a1.u[v] = *(const unsigned int*)&ldsA[wm * 32 + 16 + lr][a_k(v, half)];
    }
    AF b0, b1;
#pragma unroll
    for (int s = 0; s < 16; ++s) {
      const int kr = a_k(s >> 1, half) + (s & 1);
      b0.s[s] = ldsB[kr][wn * 32 + lr];
      b1.s[s] = ldsB[kr][wn * 32 + 16 + lr];
    }
    acc00 = wmma32f16(a0.v, b0.v, acc00);
    acc01 = wmma32f16(a0.v, b1.v, acc01);
    acc10 = wmma32f16(a1.v, b0.v, acc10);
    acc11 = wmma32f16(a1.v, b1.v, acc11);
  }

  F8 res[2][2];
  res[0][0].v = acc00; res[0][1].v = acc01;
  res[1][0].v = acc10; res[1][1].v = acc11;
#pragma unroll
  for (int mi = 0; mi < 2; ++mi)
#pragma unroll
    for (int i = 0; i < 8; ++i) {
      int M = i + half * 8;
      int rGlob = mBase + wm * 32 + mi * 16 + M;
      int n = rGlob >> 9, s = rGlob & 511;
      int b = n & 1, wb = (n >> 1) & 3, hb = n >> 3;
      int f = s >> 6, r8 = (s >> 3) & 7, c8 = s & 7;
      long obase = (((long)(b * 8 + f) * 32 + (hb * 8 + r8)) * 32 + (wb * 8 + c8)) * 512;
#pragma unroll
      for (int nt = 0; nt < 2; ++nt) {
        int c = cb + wn * 32 + nt * 16 + lr;
        out[obase + c] = (nt ? res[mi][1].f[i] : res[mi][0].f[i]) + bo[c];
      }
    }
}

// ---------------------------------------------------------------------------
// Host launcher
// ---------------------------------------------------------------------------
extern "C" void kernel_launch(void* const* d_in, const int* in_sizes, int n_in,
                              void* d_out, int out_size, void* d_ws, size_t ws_size,
                              hipStream_t stream) {
  const float* x  = (const float*)d_in[0];
  const float* Wq = (const float*)d_in[1];
  const float* Wk = (const float*)d_in[2];
  const float* Wv = (const float*)d_in[3];
  const float* Wo = (const float*)d_in[4];
  const float* bo = (const float*)d_in[5];
  float* out = (float*)d_out;

  char* ws = (char*)d_ws;
  _Float16* winh = (_Float16*)ws;                        // 16 MB; reused as Owin
  _Float16* wqh  = (_Float16*)(ws + 16777216);           // 4 x 0.5 MB weights
  _Float16* wkh  = wqh + 262144;
  _Float16* wvh  = wkh + 262144;
  _Float16* woh  = wvh + 262144;
  _Float16* Qb   = (_Float16*)(ws + 16777216 + 2097152); // 3 x 16 MB
  _Float16* Kbuf = Qb + 8388608;
  _Float16* Vbuf = Kbuf + 8388608;

  pack_win<<<dim3(8192), dim3(256), 0, stream>>>(x, winh);
  conv_w<<<dim3(256, 4), dim3(256), 0, stream>>>(Wq, Wk, Wv, Wo,
                                                 wqh, wkh, wvh, woh);
  gemm_qkv<<<dim3(128, 8), dim3(256), 0, stream>>>(winh, wqh, Qb, 1);
  gemm_qkv<<<dim3(128, 8), dim3(256), 0, stream>>>(winh, wkh, Kbuf, 1);
  gemm_qkv<<<dim3(128, 8), dim3(256), 0, stream>>>(winh, wvh, Vbuf, 0);
  // attention output aliases winh (no longer needed) as (N,S,C) f16
  attn<<<dim3(4, 256), dim3(256), 0, stream>>>(Qb, Kbuf, Vbuf, winh);
  gemm_out<<<dim3(128, 8), dim3(256), 0, stream>>>(winh, woh, bo, out);
}
